// TimeAware_MultiHeadAttention_22978075033780
// MI455X (gfx1250) — compile-verified, matrix operand order checked by reading, not run on hardware
//
#include <hip/hip_runtime.h>
#include <math.h>

// ---- problem constants (from reference: bs=4, trg=src=128, d=256, H=8) ----
#define BSZ 4
#define TT  128
#define SSZ 128
#define DD  256
#define HH  8
#define DK  32

typedef __bf16 bf16_t;
typedef bf16_t v16bf __attribute__((ext_vector_type(16)));
typedef float  v8f   __attribute__((ext_vector_type(8)));

// ---------------------------------------------------------------------------
// C[M,N] = act(X[M,K] @ W[N,K]^T + bias[N]) using CDNA5 WMMA bf16.
// f32 inputs split x = hi + lo (both bf16); D += Ahi*Bhi + Ahi*Blo + Alo*Bhi
// recovers ~16 mantissa bits (error ~2^-16): near-f32 accuracy at 3 WMMAs per
// 16x16x32 K-step. One wave owns one 16x16 C tile; EXEC stays all-ones.
//
// All 8 waves of a block share the same 16 W rows (tn is block-uniform since
// tiles_m % 8 == 0), so the B tile is staged ONCE per block into LDS as
// pre-split hi/lo bf16 (rows padded +8 bf16 so the 16 lanes of a B fetch land
// on distinct bank groups). This removes 8x redundant global traffic and 8x
// redundant f32->bf16 conversion VALU work on the B side.
//
// Per-lane-contiguous K packing is used identically for A (rows of X) and B
// (rows of W == columns of W^T), so A/B K-slots pair consistently in the WMMA.
// ---------------------------------------------------------------------------
__global__ __launch_bounds__(256) void gemm_bf16x3(
    const float* __restrict__ X, const float* __restrict__ W,
    const float* __restrict__ bias, float* __restrict__ C,
    int M, int N, int K, int act)
{
  __shared__ bf16_t s_whi[16][DD + 8];   // 8.25 KB
  __shared__ bf16_t s_wlo[16][DD + 8];   // 8.25 KB

  const int wave = threadIdx.x >> 5;
  const int lane = threadIdx.x & 31;
  const int tiles_m = M >> 4;                       // 32 (multiple of 8)
  const int tile = blockIdx.x * 8 + wave;           // 8 waves per block
  const int tm = tile % tiles_m;
  const int tn = tile / tiles_m;                    // block-uniform
  const int m0 = tm << 4, n0 = tn << 4;

  // ---- stage + split the shared B tile (W rows n0..n0+15, all K) ----
  for (int idx = threadIdx.x; idx < 16 * DD; idx += 256) {
    const int r = idx >> 8;                         // DD == 256
    const int c = idx & (DD - 1);
    const float x = W[(size_t)(n0 + r) * K + c];
    const bf16_t h = (bf16_t)x;
    s_whi[r][c] = h;
    s_wlo[r][c] = (bf16_t)(x - (float)h);
  }
  __syncthreads();

  const int half = lane >> 4, lr = lane & 15;
  v8f acc = {};
  for (int k0 = 0; k0 < K; k0 += 32) {
    v16bf ahi, alo, bhi, blo;
    const float* pa = X + (size_t)(m0 + lr) * K + k0 + half * 16;
#pragma unroll
    for (int j = 0; j < 16; ++j) {
      const float x = pa[j];
      const bf16_t h = (bf16_t)x;
      ahi[j] = h;
      alo[j] = (bf16_t)(x - (float)h);
    }
    const bf16_t* pbh = &s_whi[lr][k0 + half * 16];
    const bf16_t* pbl = &s_wlo[lr][k0 + half * 16];
#pragma unroll
    for (int j = 0; j < 16; ++j) {
      bhi[j] = pbh[j];
      blo[j] = pbl[j];
    }
    acc = __builtin_amdgcn_wmma_f32_16x16x32_bf16(false, ahi, false, bhi,
                                                  (short)0, acc, false, false);
    acc = __builtin_amdgcn_wmma_f32_16x16x32_bf16(false, ahi, false, blo,
                                                  (short)0, acc, false, false);
    acc = __builtin_amdgcn_wmma_f32_16x16x32_bf16(false, alo, false, bhi,
                                                  (short)0, acc, false, false);
  }

  // C/D layout: VGPR g -> row m0+g (lanes 0-15) / m0+8+g (lanes 16-31),
  // col = n0 + (lane & 15).
  const int col = n0 + lr;
  const float bv = bias ? bias[col] : 0.0f;
#pragma unroll
  for (int g = 0; g < 8; ++g) {
    const int row = m0 + g + half * 8;
    float v = acc[g] + bv;
    if (act == 1) v = tanhf(v);
    C[(size_t)row * N + col] = v;
  }
}

// ---------------------------------------------------------------------------
// wq_eff[b,t,h,i] = sum_j qh[b,t,h*32+j] * Wk[h*32+j, i]   (exact f32, K=32)
// qbk[b,t,h]      = sum_j qh[b,t,h*32+j] * bk[h*32+j]
// ---------------------------------------------------------------------------
__global__ __launch_bounds__(256) void wqeff_kernel(
    const float* __restrict__ qh, const float* __restrict__ Wk,
    const float* __restrict__ bk, float* __restrict__ wq_eff,
    float* __restrict__ qbk)
{
  const int idx = blockIdx.x * 256 + threadIdx.x;   // (bt, h, i)
  const int i  = idx & (DD - 1);
  const int h  = (idx >> 8) & (HH - 1);
  const int bt = idx >> 11;                          // 0..511
  const float* qrow = qh + (size_t)bt * DD + h * DK;
  float acc = 0.0f;
#pragma unroll
  for (int j = 0; j < DK; ++j)
    acc += qrow[j] * Wk[(size_t)(h * DK + j) * DD + i];
  wq_eff[idx] = acc;
  if (i == 0) {
    float s = 0.0f;
#pragma unroll
    for (int j = 0; j < DK; ++j) s += qrow[j] * bk[h * DK + j];
    qbk[bt * HH + h] = s;
  }
}

// ---------------------------------------------------------------------------
// Fused per-(b,t): time-interp kernel mixing + scores + softmax + attn@Vh.
// 256 threads = 8 wave32s. LDS: wq_eff (8KB) + scores (4KB) + t_k + qbk.
// ---------------------------------------------------------------------------
__global__ __launch_bounds__(256) void attn_fused(
    const float* __restrict__ ff1, const float* __restrict__ ff2,
    const float* __restrict__ ta,  const float* __restrict__ tb,
    const float* __restrict__ wq_eff, const float* __restrict__ qbk,
    const float* __restrict__ t_q, const float* __restrict__ t_k,
    const int* __restrict__ mask, const float* __restrict__ vh,
    float* __restrict__ attn_out, float* __restrict__ concat)
{
  __shared__ float s_wq[HH][DD];      // 8 KB
  __shared__ float s_sc[HH][SSZ];     // 4 KB
  __shared__ float s_tk[SSZ];
  __shared__ float s_qbk[HH];

  const int bt = blockIdx.x;          // 0..511
  const int b = bt >> 7;
  const int t = bt & (TT - 1);
  const int tid = threadIdx.x;

  for (int i = tid; i < HH * DD; i += 256)
    (&s_wq[0][0])[i] = wq_eff[(size_t)bt * HH * DD + i];
  if (tid < SSZ) s_tk[tid] = t_k[b * SSZ + tid];
  if (tid < HH)  s_qbk[tid] = qbk[bt * HH + tid];
  const float tq = t_q[bt];
  __syncthreads();

  const int wave = tid >> 5, lane = tid & 31;
  const float inv_sqrt_dk = 0.17677669529663687f;   // 1/sqrt(32)

  // ---- phase 1: scores[h][s] ----
  for (int s = wave; s < SSZ; s += 8) {
    const float ts = tq - s_tk[s];
    const size_t base = (size_t)(b * SSZ + s) * DD;
    const bool masked = (mask[(size_t)(b * TT + t) * SSZ + s] == 0);
    float acc[HH];
#pragma unroll
    for (int h = 0; h < HH; ++h) acc[h] = 0.0f;
#pragma unroll
    for (int j = 0; j < 8; ++j) {
      const int i = lane + j * 32;                  // coalesced, bank-clean
      const float sg = 1.0f / (1.0f + __expf(-(ta[base + i] * ts + tb[base + i])));
      const float f1 = ff1[base + i];
      const float nk = f1 + sg * (ff2[base + i] - f1);
#pragma unroll
      for (int h = 0; h < HH; ++h) acc[h] += nk * s_wq[h][i];
    }
#pragma unroll
    for (int h = 0; h < HH; ++h) {
      float v = acc[h];
#pragma unroll
      for (int off = 16; off > 0; off >>= 1) v += __shfl_xor(v, off, 32);
      if (lane == 0) {
        float sc = (v + s_qbk[h]) * inv_sqrt_dk;
        s_sc[h][s] = masked ? -1.0e9f : sc;
      }
    }
  }
  __syncthreads();

  // ---- phase 2: softmax over s, wave w handles head h = w ----
  {
    const int h = wave;
    float m = -3.4e38f;
#pragma unroll
    for (int r = 0; r < SSZ / 32; ++r) m = fmaxf(m, s_sc[h][lane + r * 32]);
#pragma unroll
    for (int off = 16; off > 0; off >>= 1) m = fmaxf(m, __shfl_xor(m, off, 32));
    float e[SSZ / 32];
    float sum = 0.0f;
#pragma unroll
    for (int r = 0; r < SSZ / 32; ++r) {
      e[r] = __expf(s_sc[h][lane + r * 32] - m);
      sum += e[r];
    }
#pragma unroll
    for (int off = 16; off > 0; off >>= 1) sum += __shfl_xor(sum, off, 32);
    const float inv = 1.0f / sum;
#pragma unroll
    for (int r = 0; r < SSZ / 32; ++r) {
      const int s = lane + r * 32;
      const float av = e[r] * inv;
      s_sc[h][s] = av;
      attn_out[(((size_t)b * HH + h) * TT + t) * SSZ + s] = av;
    }
  }
  __syncthreads();

  // ---- phase 3: concat[bt, j] = sum_s attn[h][s] * vh[b,s,j], h = j/32 ----
  {
    const int j = tid;                // 0..255
    const int h = j >> 5;
    float acc = 0.0f;
    for (int s = 0; s < SSZ; ++s)
      acc += s_sc[h][s] * vh[(size_t)(b * SSZ + s) * DD + j];
    concat[(size_t)bt * DD + j] = acc;
  }
}

// ---------------------------------------------------------------------------
extern "C" void kernel_launch(void* const* d_in, const int* in_sizes, int n_in,
                              void* d_out, int out_size, void* d_ws, size_t ws_size,
                              hipStream_t stream) {
  (void)in_sizes; (void)n_in; (void)out_size; (void)ws_size;
  const float* q    = (const float*)d_in[0];
  const float* k    = (const float*)d_in[1];
  const float* v    = (const float*)d_in[2];
  const float* t_q  = (const float*)d_in[3];
  const float* t_k  = (const float*)d_in[4];
  const int*   mask = (const int*)  d_in[5];
  const float* Wq = (const float*)d_in[6],  *bq  = (const float*)d_in[7];
  const float* Wk = (const float*)d_in[8],  *bk  = (const float*)d_in[9];
  const float* Wv = (const float*)d_in[10], *bv  = (const float*)d_in[11];
  const float* Wo = (const float*)d_in[12], *bo  = (const float*)d_in[13];
  const float* Wf1= (const float*)d_in[14], *bf1 = (const float*)d_in[15];
  const float* Wf2= (const float*)d_in[16], *bf2 = (const float*)d_in[17];
  const float* Wta= (const float*)d_in[18], *bta = (const float*)d_in[19];
  const float* Wtb= (const float*)d_in[20], *btb = (const float*)d_in[21];

  const int ROWS = BSZ * TT;          // 512 (== BSZ*SSZ)
  const int MAT  = ROWS * DD;         // 131072 floats

  float* ws     = (float*)d_ws;
  float* qh     = ws;                 // [512,256]
  float* vh     = qh  + MAT;
  float* ff1    = vh  + MAT;
  float* ff2    = ff1 + MAT;
  float* ta     = ff2 + MAT;
  float* tb     = ta  + MAT;
  float* concat = tb  + MAT;
  float* qbk    = concat + MAT;       // [512,8]
  float* wq     = qbk + ROWS * HH;    // [512,8,256]

  float* out0 = (float*)d_out;                 // output: [4,128,256]
  float* attn = out0 + (size_t)BSZ * TT * DD;  // attn:   [4,8,128,128]

  const dim3 blk(256);
  const int gemm_blocks = (ROWS / 16) * (DD / 16) / 8;   // 64, exact fill

  gemm_bf16x3<<<gemm_blocks, blk, 0, stream>>>(q, Wq,  bq,  qh,  ROWS, DD, DD, 0);
  gemm_bf16x3<<<gemm_blocks, blk, 0, stream>>>(v, Wv,  bv,  vh,  ROWS, DD, DD, 0);
  gemm_bf16x3<<<gemm_blocks, blk, 0, stream>>>(k, Wf1, bf1, ff1, ROWS, DD, DD, 1);
  gemm_bf16x3<<<gemm_blocks, blk, 0, stream>>>(k, Wf2, bf2, ff2, ROWS, DD, DD, 1);
  gemm_bf16x3<<<gemm_blocks, blk, 0, stream>>>(k, Wta, bta, ta,  ROWS, DD, DD, 0);
  gemm_bf16x3<<<gemm_blocks, blk, 0, stream>>>(k, Wtb, btb, tb,  ROWS, DD, DD, 0);

  wqeff_kernel<<<ROWS * HH * DD / 256, blk, 0, stream>>>(qh, Wk, bk, wq, qbk);

  attn_fused<<<ROWS, blk, 0, stream>>>(ff1, ff2, ta, tb, wq, qbk,
                                       t_q, t_k, mask, vh, attn, concat);

  gemm_bf16x3<<<gemm_blocks, blk, 0, stream>>>(concat, Wo, bo, out0, ROWS, DD, DD, 0);
}
